// LSTMClassificationModel_90211493085285
// MI455X (gfx1250) — matrix operand order repaired and checked by reading, not compiled, not run on hardware
//
#include <hip/hip_runtime.h>
#include <hip/hip_bf16.h>
#include <stdint.h>

#define TT 512
#define BB 512
#define DD 128
#define HH 100
#define G4 400
#define NTILE 25          // 400 / 16 column tiles
#define ZSTR 404          // padded Z row stride (floats) to avoid bank conflicts

typedef __attribute__((ext_vector_type(16))) __bf16 v16bf;
typedef __attribute__((ext_vector_type(8)))  float  v8f;

__device__ __forceinline__ unsigned short f2bf(float f) {
  unsigned u = __float_as_uint(f);
  u += 0x7fffu + ((u >> 16) & 1u);          // round-to-nearest-even
  return (unsigned short)(u >> 16);
}
__device__ __forceinline__ __bf16 f2bfh(float f) {
  unsigned short s = f2bf(f);
  return __builtin_bit_cast(__bf16, s);
}
__device__ __forceinline__ float sigmoidf_(float x) {
  return 1.0f / (1.0f + __expf(-x));
}

// Build one 32x16 bf16 B-fragment (for v_wmma_f32_16x16x32_bf16) straight from
// the row-major f32 weight matrix in global memory (L2-resident; startup only).
// Layout (ISA 7.12.2): lane = hl*16 + n (hl = lane half, n = column), half j2 -> K = hl*8 + (j2&7) + (j2>>3)*16.
__device__ __forceinline__ v16bf load_bfrag(const float* __restrict__ W, int kmax,
                                            int kb, int nb, int lane) {
  const int n     = nb * 16 + (lane & 15);
  const int kbase = kb * 32 + ((lane >> 4) << 3);
  v16bf f;
#pragma unroll
  for (int j = 0; j < 16; ++j) {
    const int k = kbase + (j & 7) + ((j >> 3) << 4);
    const float v = (k < kmax) ? W[k * G4 + n] : 0.0f;
    f[j] = f2bfh(v);
  }
  return f;
}

extern "C" __global__ void __launch_bounds__(256)
lstm_fused_kernel(const int*   __restrict__ tokens,
                  const float* __restrict__ emb,
                  const float* __restrict__ Wx,      // [D, 4H]
                  const float* __restrict__ Wr,      // [H, 4H]
                  const float* __restrict__ bias,    // [4H]
                  const float* __restrict__ Wd,      // [H, NCLS]
                  const float* __restrict__ bd,      // [NCLS]
                  float*       __restrict__ out)     // [B, NCLS]
{
  // LDS: A-fragments (shared by all 8 waves), x staging for async copy,
  // Z buffer, c-state, final h (f32). ~55 KB total.
  __shared__ __align__(32) unsigned short XAF[4 * 512];   // x A-frags (16x128 bf16)
  __shared__ __align__(32) unsigned short HAF[4 * 512];   // h A-frags (16x128 bf16, K>=100 zero)
  __shared__ __align__(16) float XSTG[16 * 128];          // async-gathered f32 emb rows
  __shared__ float ZB[16 * ZSTR];                         // pre-activation gates (f32)
  __shared__ float CST[16 * HH];                          // cell state
  __shared__ float HFP[16 * HH];                          // final hidden (f32)

  const int tid  = threadIdx.x;
  const int lane = tid & 31;
  const int wave = tid >> 5;
  const int b0   = blockIdx.x * 16;

  // Column-tile slots for this wave: nb = wave, wave+8, wave+16, wave+24.
  int  nbv[4];
  bool stv[4];
#pragma unroll
  for (int s = 0; s < 4; ++s) {
    const int nb = wave + 8 * s;
    stv[s] = (nb < NTILE);
    nbv[s] = stv[s] ? nb : (NTILE - 1);     // clamp: compute dup, skip store
  }

  // Pin both weight matrices in VGPRs as pre-swizzled B-fragments (startup only).
  v16bf wxr[4][4], wrr[4][4];
  float bv[4];
#pragma unroll
  for (int s = 0; s < 4; ++s) {
    bv[s] = bias[nbv[s] * 16 + (lane & 15)];
#pragma unroll
    for (int kb = 0; kb < 4; ++kb) {
      wxr[s][kb] = load_bfrag(Wx, DD, kb, nbv[s], lane);
      wrr[s][kb] = load_bfrag(Wr, HH, kb, nbv[s], lane);   // rows 100..127 -> 0
    }
  }

  // Zero h-fragments (h0 = 0; also keeps K=100..127 padding zero forever) and c.
  for (int i = tid; i < 4 * 512; i += 256) HAF[i] = 0;
  for (int i = tid; i < 16 * HH; i += 256) CST[i] = 0.0f;

  const unsigned xstg_base = (unsigned)(unsigned long long)(void*)&XSTG[0];

  // Prefetch embeddings for t = 0: 16 rows x 512 B via async B128 copies to LDS.
#pragma unroll
  for (int it = 0; it < 2; ++it) {
    const int chunk = tid + it * 256;                 // 512 chunks of 16 B
    const int r     = chunk >> 5;
    const int offb  = (chunk & 31) << 4;
    const int tok   = tokens[(b0 + r) * TT + 0];
    const char* g   = (const char*)emb + ((unsigned long long)tok << 9) + offb;
    const unsigned la = xstg_base + (unsigned)(r * 512 + offb);
    asm volatile("global_load_async_to_lds_b128 %0, %1, off" :: "v"(la), "v"(g) : "memory");
  }

#pragma unroll 1
  for (int t = 0; t < TT; ++t) {
    asm volatile("s_wait_asynccnt 0x0" ::: "memory");
    __syncthreads();                                  // XSTG(t) ready, HAF(t) ready

    // Scatter-convert f32 x rows into bf16 A-fragment layout.
#pragma unroll
    for (int it = 0; it < 8; ++it) {
      const int e = tid + it * 256;                   // 0..2047
      const int r = e >> 7, k = e & 127;
      const float v = XSTG[e];
      const int kb = k >> 5, kl = k & 31;
      const int hl = (kl >> 3) & 1;
      const int j2 = ((kl >> 4) << 3) | (kl & 7);
      const int l  = (hl << 4) | r;
      XAF[kb * 512 + l * 16 + j2] = f2bf(v);
    }
    __syncthreads();                                  // XAF ready, XSTG free

    // Prefetch step t+1's embeddings (overlaps with WMMA + gate math below).
    if (t + 1 < TT) {
#pragma unroll
      for (int it = 0; it < 2; ++it) {
        const int chunk = tid + it * 256;
        const int r     = chunk >> 5;
        const int offb  = (chunk & 31) << 4;
        const int tok   = tokens[(b0 + r) * TT + (t + 1)];
        const char* g   = (const char*)emb + ((unsigned long long)tok << 9) + offb;
        const unsigned la = xstg_base + (unsigned)(r * 512 + offb);
        asm volatile("global_load_async_to_lds_b128 %0, %1, off" :: "v"(la), "v"(g) : "memory");
      }
    }

    // Z = X*Wx + H*Wr + b   (200 WMMAs / WG / step, weights already in VGPRs)
    v16bf xa[4], ha[4];
#pragma unroll
    for (int kb = 0; kb < 4; ++kb) {
      xa[kb] = *(const v16bf*)(&XAF[kb * 512 + lane * 16]);
      ha[kb] = *(const v16bf*)(&HAF[kb * 512 + lane * 16]);
    }
#pragma unroll
    for (int s = 0; s < 4; ++s) {
      const float b = bv[s];
      v8f acc = {b, b, b, b, b, b, b, b};
#pragma unroll
      for (int kb = 0; kb < 4; ++kb)
        acc = __builtin_amdgcn_wmma_f32_16x16x32_bf16(false, xa[kb], false, wxr[s][kb],
                                                      (short)0, acc, false, false);
#pragma unroll
      for (int kb = 0; kb < 4; ++kb)
        acc = __builtin_amdgcn_wmma_f32_16x16x32_bf16(false, ha[kb], false, wrr[s][kb],
                                                      (short)0, acc, false, false);
      if (stv[s]) {
        const int col = nbv[s] * 16 + (lane & 15);
        const int rb  = (lane >> 4) * 8;
#pragma unroll
        for (int r = 0; r < 8; ++r) ZB[(rb + r) * ZSTR + col] = acc[r];
      }
    }
    __syncthreads();                                  // Z ready

    // Gates + state update; write h straight into A-fragment layout.
#pragma unroll 1
    for (int e = tid; e < 16 * HH; e += 256) {
      const int r = e / HH, n = e - r * HH;
      const float* zr = &ZB[r * ZSTR];
      const float ig = sigmoidf_(zr[n]);
      const float fg = sigmoidf_(zr[100 + n]);
      const float gg = tanhf(zr[200 + n]);
      const float og = sigmoidf_(zr[300 + n]);
      const float c  = fg * CST[e] + ig * gg;
      CST[e] = c;
      const float h = og * tanhf(c);
      const int kb = n >> 5, kl = n & 31;
      const int hl = (kl >> 3) & 1;
      const int j2 = ((kl >> 4) << 3) | (kl & 7);
      const int l  = (hl << 4) | r;
      HAF[kb * 512 + l * 16 + j2] = f2bf(h);
      if (t == TT - 1) HFP[e] = h;
    }
    __syncthreads();                                  // HAF(t+1) ready, Z free
  }

  // Dense [16,100]x[100,4] + softmax (tiny; scalar is fine).
  if (tid < 64) {
    const int r = tid >> 2, c = tid & 3;
    float a = bd[c];
#pragma unroll 4
    for (int k = 0; k < HH; ++k) a += HFP[r * HH + k] * Wd[k * 4 + c];
    ZB[r * 4 + c] = a;
  }
  __syncthreads();
  if (tid < 16) {
    const int r = tid;
    const float l0 = ZB[r * 4 + 0], l1 = ZB[r * 4 + 1];
    const float l2 = ZB[r * 4 + 2], l3 = ZB[r * 4 + 3];
    const float m  = fmaxf(fmaxf(l0, l1), fmaxf(l2, l3));
    const float e0 = __expf(l0 - m), e1 = __expf(l1 - m);
    const float e2 = __expf(l2 - m), e3 = __expf(l3 - m);
    const float inv = 1.0f / (e0 + e1 + e2 + e3);
    float* o = out + (b0 + r) * 4;
    o[0] = e0 * inv; o[1] = e1 * inv; o[2] = e2 * inv; o[3] = e3 * inv;
  }
}

extern "C" void kernel_launch(void* const* d_in, const int* in_sizes, int n_in,
                              void* d_out, int out_size, void* d_ws, size_t ws_size,
                              hipStream_t stream) {
  (void)in_sizes; (void)n_in; (void)out_size; (void)d_ws; (void)ws_size;
  const int*   tokens = (const int*)d_in[0];
  const float* emb    = (const float*)d_in[1];
  const float* Wx     = (const float*)d_in[2];
  const float* Wr     = (const float*)d_in[3];
  const float* bias   = (const float*)d_in[4];
  const float* Wd     = (const float*)d_in[5];
  const float* bd     = (const float*)d_in[6];
  lstm_fused_kernel<<<dim3(BB / 16), dim3(256), 0, stream>>>(
      tokens, emb, Wx, Wr, bias, Wd, bd, (float*)d_out);
}
